// LinearSNN_15831249453626
// MI455X (gfx1250) — compile-verified, matrix operand order checked by reading, not run on hardware
//
#include <hip/hip_runtime.h>
#include <hip/hip_bf16.h>

// ---------------- CDNA5 WMMA types ----------------
typedef __bf16 bf16_t;
typedef bf16_t v16bf __attribute__((ext_vector_type(16)));
typedef float  v8f   __attribute__((ext_vector_type(8)));
typedef float        f32x4 __attribute__((ext_vector_type(4)));
typedef unsigned int u32x2 __attribute__((ext_vector_type(2)));
typedef unsigned int u32x4 __attribute__((ext_vector_type(4)));

union BFrag { unsigned int u[8]; u32x4 q[2]; v16bf v; };

#define B_   256
#define T_   250
#define CIN  700
#define H_   512
#define O_   20

#define WMMA_BF16(A, Bm, C) __builtin_amdgcn_wmma_f32_16x16x32_bf16( \
    false, (A).v, false, (Bm).v, (short)0, (C), false, false)

// float -> bf16 (round to nearest even)
__device__ __forceinline__ unsigned short f2bf(float f) {
    unsigned int u = __builtin_bit_cast(unsigned int, f);
    u += 0x7fffu + ((u >> 16) & 1u);
    return (unsigned short)(u >> 16);
}
__device__ __forceinline__ float bf2f(unsigned short h) {
    unsigned int u = ((unsigned int)h) << 16;
    return __builtin_bit_cast(float, u);
}
__device__ __forceinline__ unsigned int pack2(float a, float b) {
    return (unsigned int)f2bf(a) | ((unsigned int)f2bf(b) << 16);
}
// Build A fragment from two contiguous 8-float runs (ISA K-interleave:
// lane half selects K+8 block; VGPR j<4 covers run0, j>=4 covers run1).
__device__ __forceinline__ void packA(BFrag& A, f32x4 a0, f32x4 a1,
                                      f32x4 b0, f32x4 b1) {
    A.u[0] = pack2(a0.x, a0.y); A.u[1] = pack2(a0.z, a0.w);
    A.u[2] = pack2(a1.x, a1.y); A.u[3] = pack2(a1.z, a1.w);
    A.u[4] = pack2(b0.x, b0.y); A.u[5] = pack2(b0.z, b0.w);
    A.u[6] = pack2(b1.x, b1.y); A.u[7] = pack2(b1.z, b1.w);
}
// B fragment from bf16 row with stride-700 rows (only 8B alignment): 4x b64
__device__ __forceinline__ void loadB64(BFrag& Bm, const unsigned short* p) {
    u32x2 a = *(const u32x2*)(p);
    u32x2 b = *(const u32x2*)(p + 4);
    u32x2 c = *(const u32x2*)(p + 16);
    u32x2 d = *(const u32x2*)(p + 20);
    Bm.u[0] = a.x; Bm.u[1] = a.y; Bm.u[2] = b.x; Bm.u[3] = b.y;
    Bm.u[4] = c.x; Bm.u[5] = c.y; Bm.u[6] = d.x; Bm.u[7] = d.y;
}
// B fragment from bf16 row with 16B-aligned runs: 2x b128
__device__ __forceinline__ void loadB128(BFrag& Bm, const unsigned short* p) {
    Bm.q[0] = *(const u32x4*)(p);
    Bm.q[1] = *(const u32x4*)(p + 16);
}

// ---------------- Phase A: cast weights fp32 -> bf16 ----------------
__global__ void snn_cast_bf16(const float* __restrict__ src,
                              unsigned short* __restrict__ dst, int n) {
    int i = blockIdx.x * blockDim.x + threadIdx.x;
    if (i < n) dst[i] = f2bf(src[i]);
}

// ---------------- Phase B: i_ff[t][b][h] = sum_c x[b][t][c] * w1[h][c] ----
// Block = 256 threads = 8 waves; block owns 16 rows (r = t*256+b) x 512 cols.
// K = 700 = 21 full 32-tiles + tail tile (handled separately, branch-light).
__global__ __launch_bounds__(256) void snn_ff_gemm(
    const float* __restrict__ x, const unsigned short* __restrict__ w1b,
    float* __restrict__ iff)
{
    const int tid  = threadIdx.x;
    const int wave = tid >> 5;
    const int lane = tid & 31;
    const int l15  = lane & 15;
    const int half = lane >> 4;
    const int hk   = 8 * half;
    const int rowbase = blockIdx.x * 16;

    const int r = rowbase + l15;              // r = t*256 + b
    const int t = r >> 8;
    const int b = r & 255;
    const float* xrow = x + ((size_t)b * T_ + t) * CIN;
    const int colw = wave * 64;

    const unsigned short* w0 = w1b + (size_t)(colw +  0 + l15) * CIN;
    const unsigned short* w1r = w1b + (size_t)(colw + 16 + l15) * CIN;
    const unsigned short* w2 = w1b + (size_t)(colw + 32 + l15) * CIN;
    const unsigned short* w3 = w1b + (size_t)(colw + 48 + l15) * CIN;

    v8f acc0 = {}, acc1 = {}, acc2 = {}, acc3 = {};

    for (int kt = 0; kt < 21; ++kt) {
        const int kb = kt * 32 + hk;          // run0: kb..kb+7, run1: kb+16..+23
        BFrag A, B0, B1, B2, B3;
        packA(A, *(const f32x4*)(xrow + kb),      *(const f32x4*)(xrow + kb + 4),
                 *(const f32x4*)(xrow + kb + 16), *(const f32x4*)(xrow + kb + 20));
        loadB64(B0, w0 + kb); loadB64(B1, w1r + kb);
        loadB64(B2, w2 + kb); loadB64(B3, w3 + kb);
        acc0 = WMMA_BF16(A, B0, acc0);
        acc1 = WMMA_BF16(A, B1, acc1);
        acc2 = WMMA_BF16(A, B2, acc2);
        acc3 = WMMA_BF16(A, B3, acc3);
    }
    {   // tail tile kt=21: run0 = 672+hk (all valid), run1 = 688+hk
        const int kb = 672 + hk;
        f32x4 fa0 = *(const f32x4*)(xrow + kb);        // <= 687
        f32x4 fa1 = *(const f32x4*)(xrow + kb + 4);
        f32x4 fb0 = *(const f32x4*)(xrow + kb + 16);   // 688..691 / 696..699
        f32x4 fb1 = {};                                // 692..695 valid half==0 only
        if (half == 0) fb1 = *(const f32x4*)(xrow + kb + 20);
        BFrag A, B0, B1, B2, B3;
        packA(A, fa0, fa1, fb0, fb1);
        u32x2 z2; z2.x = 0u; z2.y = 0u;
        u32x2 t0 = z2, t1 = z2, t2 = z2, t3 = z2;
        if (half == 0) {
            t0 = *(const u32x2*)(w0 + kb + 20);  t1 = *(const u32x2*)(w1r + kb + 20);
            t2 = *(const u32x2*)(w2 + kb + 20);  t3 = *(const u32x2*)(w3 + kb + 20);
        }
        BFrag* bp[1];
        (void)bp;
        // run0 + first half of run1 are always in-bounds
        {
            u32x2 a = *(const u32x2*)(w0 + kb);   u32x2 b2 = *(const u32x2*)(w0 + kb + 4);
            u32x2 c = *(const u32x2*)(w0 + kb + 16);
            B0.u[0]=a.x; B0.u[1]=a.y; B0.u[2]=b2.x; B0.u[3]=b2.y;
            B0.u[4]=c.x; B0.u[5]=c.y; B0.u[6]=t0.x; B0.u[7]=t0.y;
        }
        {
            u32x2 a = *(const u32x2*)(w1r + kb);  u32x2 b2 = *(const u32x2*)(w1r + kb + 4);
            u32x2 c = *(const u32x2*)(w1r + kb + 16);
            B1.u[0]=a.x; B1.u[1]=a.y; B1.u[2]=b2.x; B1.u[3]=b2.y;
            B1.u[4]=c.x; B1.u[5]=c.y; B1.u[6]=t1.x; B1.u[7]=t1.y;
        }
        {
            u32x2 a = *(const u32x2*)(w2 + kb);   u32x2 b2 = *(const u32x2*)(w2 + kb + 4);
            u32x2 c = *(const u32x2*)(w2 + kb + 16);
            B2.u[0]=a.x; B2.u[1]=a.y; B2.u[2]=b2.x; B2.u[3]=b2.y;
            B2.u[4]=c.x; B2.u[5]=c.y; B2.u[6]=t2.x; B2.u[7]=t2.y;
        }
        {
            u32x2 a = *(const u32x2*)(w3 + kb);   u32x2 b2 = *(const u32x2*)(w3 + kb + 4);
            u32x2 c = *(const u32x2*)(w3 + kb + 16);
            B3.u[0]=a.x; B3.u[1]=a.y; B3.u[2]=b2.x; B3.u[3]=b2.y;
            B3.u[4]=c.x; B3.u[5]=c.y; B3.u[6]=t3.x; B3.u[7]=t3.y;
        }
        acc0 = WMMA_BF16(A, B0, acc0);
        acc1 = WMMA_BF16(A, B1, acc1);
        acc2 = WMMA_BF16(A, B2, acc2);
        acc3 = WMMA_BF16(A, B3, acc3);
    }
    // D layout: VGPR vr -> row vr + 8*half, col = l15 within tile
    float* orow = iff + (size_t)rowbase * H_ + colw + l15;
#pragma unroll
    for (int vr = 0; vr < 8; ++vr) {
        float* p = orow + (size_t)(vr + 8 * half) * H_;
        p[0]  = acc0[vr];
        p[16] = acc1[vr];
        p[32] = acc2[vr];
        p[48] = acc3[vr];
    }
}

// ---------------- Phase C: sequential LIF scan ----------------
__global__ __launch_bounds__(256) void snn_scan(
    const float* __restrict__ iff,
    const unsigned short* __restrict__ wrecb,
    const float* __restrict__ w_out,
    const float* __restrict__ alpha, const float* __restrict__ rho,
    const float* __restrict__ beta_a, const float* __restrict__ beta_out,
    float* __restrict__ out)
{
    __shared__ unsigned short s_lds[16 * H_];   // 16 KB spikes (bf16)
    __shared__ float wout_lds[O_ * H_];         // 40 KB readout weights

    const int tid  = threadIdx.x;
    const int wave = tid >> 5;
    const int lane = tid & 31;
    const int l15  = lane & 15;
    const int half = lane >> 4;
    const int hk   = 8 * half;
    const int rowbase = blockIdx.x * 16;        // batch rows
    const int colw = wave * 64;

    for (int i = tid; i < O_ * H_; i += 256) wout_lds[i] = w_out[i];
    for (int i = tid; i < 16 * H_; i += 256) s_lds[i] = 0;

    float al[4], oma[4], rh[4], ba[4];
#pragma unroll
    for (int n = 0; n < 4; ++n) {
        int col = colw + n * 16 + l15;
        al[n] = alpha[col]; oma[n] = 1.f - al[n];
        rh[n] = rho[col];   ba[n]  = beta_a[col];
    }

    const unsigned short* wr0 = wrecb + (size_t)(colw +  0 + l15) * H_;
    const unsigned short* wr1 = wrecb + (size_t)(colw + 16 + l15) * H_;
    const unsigned short* wr2 = wrecb + (size_t)(colw + 32 + l15) * H_;
    const unsigned short* wr3 = wrecb + (size_t)(colw + 48 + l15) * H_;
    const unsigned short* srowA = &s_lds[l15 * H_];

    v8f v1[4] = {};
    v8f a1[4] = {};
    v8f sp[4] = {};                             // previous spikes (D layout)

    const int p0 = tid, p1 = tid + 256;
    float vo0 = 0.f, os0 = 0.f, vo1 = 0.f, os1 = 0.f;
    const float bo0 = beta_out[p0 % O_];
    const float bo1 = (p1 < 16 * O_) ? beta_out[p1 % O_] : 0.f;

    __syncthreads();

    for (int t = 0; t < T_; ++t) {
        // recurrent GEMM: i1 = s_prev @ w_rec^T  (16 x 512 x 512)
        v8f acc[4] = {};
        for (int kt = 0; kt < 16; ++kt) {
            const int kb = kt * 32 + hk;
            BFrag A, BF0, BF1, BF2, BF3;
            A.q[0] = *(const u32x4*)(srowA + kb);        // ds_load_b128
            A.q[1] = *(const u32x4*)(srowA + kb + 16);
            loadB128(BF0, wr0 + kb);
            loadB128(BF1, wr1 + kb);
            loadB128(BF2, wr2 + kb);
            loadB128(BF3, wr3 + kb);
            acc[0] = WMMA_BF16(A, BF0, acc[0]);
            acc[1] = WMMA_BF16(A, BF1, acc[1]);
            acc[2] = WMMA_BF16(A, BF2, acc[2]);
            acc[3] = WMMA_BF16(A, BF3, acc[3]);
        }
        __syncthreads();                        // all reads of old spikes done

        // adaptive-LIF elementwise update (state lives in registers)
        const float* ifft = iff + (size_t)t * (B_ * H_);
#pragma unroll
        for (int n = 0; n < 4; ++n) {
            const int col = colw + n * 16 + l15;
#pragma unroll
            for (int vr = 0; vr < 8; ++vr) {
                int m = vr + 8 * half;
                float i1 = acc[n][vr] + ifft[(size_t)(rowbase + m) * H_ + col];
                float a  = rh[n] * a1[n][vr] + ba[n] * sp[n][vr];
                float v  = al[n] * v1[n][vr] + oma[n] * i1 - a;
                float s  = (v > 1.0f) ? 1.f : 0.f;   // spike(v - THRESH)
                v -= s;                               // reset by subtraction
                a1[n][vr] = a; v1[n][vr] = v; sp[n][vr] = s;
                s_lds[m * H_ + col] = f2bf(s);        // exact for {0,1}
            }
        }
        __syncthreads();                        // new spikes visible

        // readout LIF: i_out = s1 @ w_out^T (16 x 512 x 20), from LDS
        {
            int bl = p0 / O_, o = p0 % O_;
            const unsigned int* srow = (const unsigned int*)&s_lds[bl * H_];
            const float* wr = &wout_lds[o * H_];
            float d = 0.f;
#pragma unroll 4
            for (int h2 = 0; h2 < H_ / 2; ++h2) {
                unsigned int u = srow[h2];
                d += bf2f((unsigned short)u)         * wr[2 * h2];
                d += bf2f((unsigned short)(u >> 16)) * wr[2 * h2 + 1];
            }
            float vo = bo0 * vo0 + (1.f - bo0) * d;
            float so = (vo > 1.f) ? 1.f : 0.f;
            vo -= so; vo0 = vo; os0 += vo;
        }
        if (p1 < 16 * O_) {
            int bl = p1 / O_, o = p1 % O_;
            const unsigned int* srow = (const unsigned int*)&s_lds[bl * H_];
            const float* wr = &wout_lds[o * H_];
            float d = 0.f;
#pragma unroll 4
            for (int h2 = 0; h2 < H_ / 2; ++h2) {
                unsigned int u = srow[h2];
                d += bf2f((unsigned short)u)         * wr[2 * h2];
                d += bf2f((unsigned short)(u >> 16)) * wr[2 * h2 + 1];
            }
            float vo = bo1 * vo1 + (1.f - bo1) * d;
            float so = (vo > 1.f) ? 1.f : 0.f;
            vo -= so; vo1 = vo; os1 += vo;
        }
    }

    // out[b][o] = out_sum / T
    out[(size_t)(rowbase + p0 / O_) * O_ + (p0 % O_)] = os0 * (1.f / T_);
    if (p1 < 16 * O_)
        out[(size_t)(rowbase + p1 / O_) * O_ + (p1 % O_)] = os1 * (1.f / T_);
}

// ---------------- host launcher ----------------
extern "C" void kernel_launch(void* const* d_in, const int* in_sizes, int n_in,
                              void* d_out, int out_size, void* d_ws, size_t ws_size,
                              hipStream_t stream) {
    const float* x        = (const float*)d_in[0];   // [256,250,700]
    const float* w1       = (const float*)d_in[1];   // [512,700]
    const float* w_rec    = (const float*)d_in[2];   // [512,512]
    const float* w_out    = (const float*)d_in[3];   // [20,512]
    const float* alpha    = (const float*)d_in[4];
    const float* rho      = (const float*)d_in[5];
    const float* beta_a   = (const float*)d_in[6];
    const float* beta_out = (const float*)d_in[7];
    float* out = (float*)d_out;

    char* ws = (char*)d_ws;
    const size_t off_w1b  = 0;                       // 512*700*2 = 716800 B
    const size_t off_wrec = 716800;                  // 512*512*2 = 524288 B
    const size_t off_iff  = 716800 + 524288;         // 250*256*512*4 B
    unsigned short* w1b   = (unsigned short*)(ws + off_w1b);
    unsigned short* wrecb = (unsigned short*)(ws + off_wrec);
    float*          iff   = (float*)(ws + off_iff);

    snn_cast_bf16<<<(H_ * CIN + 255) / 256, 256, 0, stream>>>(w1, w1b, H_ * CIN);
    snn_cast_bf16<<<(H_ * H_  + 255) / 256, 256, 0, stream>>>(w_rec, wrecb, H_ * H_);
    snn_ff_gemm<<<(T_ * B_) / 16, 256, 0, stream>>>(x, w1b, iff);
    snn_scan<<<B_ / 16, 256, 0, stream>>>(iff, wrecb, w_out, alpha, rho,
                                          beta_a, beta_out, out);
}
